// Model_SwinT_84043920048313
// MI455X (gfx1250) — matrix operand order repaired and checked by reading, not compiled, not run on hardware
//
#include <hip/hip_runtime.h>
#include <hip/hip_bf16.h>
#include <math.h>

typedef __bf16 bf16_t;
typedef __attribute__((ext_vector_type(16))) __bf16 v16bf_t;
typedef __attribute__((ext_vector_type(8)))  __bf16 v8bf_t;
typedef __attribute__((ext_vector_type(8)))  float  v8f_t;

union AFrag { v16bf_t v; v8bf_t h[2]; };

// ---------------------------------------------------------------------------
// f32 -> bf16 convert (weights, pooled activations)
// ---------------------------------------------------------------------------
__global__ void k_f32_to_bf16(const float* __restrict__ in, bf16_t* __restrict__ out, long n)
{
    long i = (long)blockIdx.x * blockDim.x + threadIdx.x;
    if (i < n) out[i] = (bf16_t)in[i];
}

// ---------------------------------------------------------------------------
// Patch embed: 4x4/4 conv, NCHW in -> NHWC out (+bias).  x:(B,3,224,224) w:(96,3,4,4)
// ---------------------------------------------------------------------------
__global__ void k_patch_embed(const float* __restrict__ x, const float* __restrict__ w,
                              const float* __restrict__ bias, float* __restrict__ out, long total)
{
    long t = (long)blockIdx.x * blockDim.x + threadIdx.x;
    if (t >= total) return;
    int oc = (int)(t % 96); long r = t / 96;
    int ow = (int)(r % 56); r /= 56;
    int oh = (int)(r % 56); int b = (int)(r / 56);
    float acc = bias[oc];
    for (int ci = 0; ci < 3; ++ci)
        for (int kh = 0; kh < 4; ++kh)
            for (int kw = 0; kw < 4; ++kw)
                acc += x[(((size_t)b * 3 + ci) * 224 + (oh * 4 + kh)) * 224 + (ow * 4 + kw)]
                     * w[(((size_t)oc * 3 + ci) * 4 + kh) * 4 + kw];
    out[t] = acc;
}

// ---------------------------------------------------------------------------
// LayerNorm over last dim C (one block per row).  In-place safe for f32 out.
// ---------------------------------------------------------------------------
template<bool OUT_BF16>
__global__ __launch_bounds__(128)
void k_layernorm(const float* __restrict__ x, const float* __restrict__ w,
                 const float* __restrict__ b, void* __restrict__ out, int C)
{
    __shared__ float red[128];
    size_t row = blockIdx.x;
    const float* xr = x + row * (size_t)C;
    int tid = threadIdx.x;
    float s = 0.f;
    for (int c = tid; c < C; c += 128) s += xr[c];
    red[tid] = s; __syncthreads();
    for (int o = 64; o > 0; o >>= 1) { if (tid < o) red[tid] += red[tid + o]; __syncthreads(); }
    float mean = red[0] / C;
    __syncthreads();
    float v = 0.f;
    for (int c = tid; c < C; c += 128) { float d = xr[c] - mean; v += d * d; }
    red[tid] = v; __syncthreads();
    for (int o = 64; o > 0; o >>= 1) { if (tid < o) red[tid] += red[tid + o]; __syncthreads(); }
    float rstd = rsqrtf(red[0] / C + 1e-5f);
    for (int c = tid; c < C; c += 128) {
        float y = (xr[c] - mean) * rstd * w[c] + b[c];
        if (OUT_BF16) ((bf16_t*)out)[row * (size_t)C + c] = (bf16_t)y;
        else          ((float*)out)[row * (size_t)C + c] = y;
    }
}

// ---------------------------------------------------------------------------
// WMMA bf16 GEMM: out = A(MxK) * W(NxK)^T + bias [+GELU] [+=residual]
// One wave per 32x32 output tile (2x2 WMMA tiles, 4 accumulators), 4 waves/blk.
// Per K-step: 8 x 16B loads feed 4 WMMAs.  M%32==0, N%32==0, K%32==0.
// ---------------------------------------------------------------------------
template<bool OUT_BF16, bool GELU, bool ACCUM>
__global__ __launch_bounds__(128)
void k_gemm(const bf16_t* __restrict__ A, const bf16_t* __restrict__ W,
            const float* __restrict__ bias, void* __restrict__ out,
            int M, int N, int K)
{
    int wave = threadIdx.x >> 5;
    int lane = threadIdx.x & 31;
    int nt = blockIdx.x;                 // N/32 super-tile
    int mt = blockIdx.y * 4 + wave;      // M/32 super-tile
    if (mt * 32 >= M) return;
    int l15 = lane & 15;
    int hi  = lane >> 4;
    const bf16_t* arow0 = A + (size_t)(mt * 32 + l15) * K + hi * 8;   // A: K = base..+7, base+16..+23
    const bf16_t* arow1 = arow0 + (size_t)16 * K;
    const bf16_t* brow0 = W + (size_t)(nt * 32 + l15) * K + hi * 8;   // B: K = base..+7, base+16..+23
    const bf16_t* brow1 = brow0 + (size_t)16 * K;
    v8f_t acc00 = {}, acc01 = {}, acc10 = {}, acc11 = {};
    for (int k0 = 0; k0 < K; k0 += 32) {
        __builtin_prefetch(arow0 + k0 + 64, 0, 3);   // global_prefetch_b8
        __builtin_prefetch(brow0 + k0 + 64, 0, 3);
        AFrag a0, a1, b0, b1;
        a0.h[0] = *(const v8bf_t*)(arow0 + k0);
        a0.h[1] = *(const v8bf_t*)(arow0 + k0 + 16);
        a1.h[0] = *(const v8bf_t*)(arow1 + k0);
        a1.h[1] = *(const v8bf_t*)(arow1 + k0 + 16);
        b0.h[0] = *(const v8bf_t*)(brow0 + k0);
        b0.h[1] = *(const v8bf_t*)(brow0 + k0 + 16);
        b1.h[0] = *(const v8bf_t*)(brow1 + k0);
        b1.h[1] = *(const v8bf_t*)(brow1 + k0 + 16);
        acc00 = __builtin_amdgcn_wmma_f32_16x16x32_bf16(false, a0.v, false, b0.v,
                                                        (short)0, acc00, false, false);
        acc01 = __builtin_amdgcn_wmma_f32_16x16x32_bf16(false, a0.v, false, b1.v,
                                                        (short)0, acc01, false, false);
        acc10 = __builtin_amdgcn_wmma_f32_16x16x32_bf16(false, a1.v, false, b0.v,
                                                        (short)0, acc10, false, false);
        acc11 = __builtin_amdgcn_wmma_f32_16x16x32_bf16(false, a1.v, false, b1.v,
                                                        (short)0, acc11, false, false);
    }
    v8f_t accs[2][2] = { { acc00, acc01 }, { acc10, acc11 } };
#pragma unroll
    for (int mh = 0; mh < 2; ++mh) {
#pragma unroll
        for (int nh = 0; nh < 2; ++nh) {
            int n = nt * 32 + nh * 16 + l15;
            float bv = bias ? bias[n] : 0.0f;
            v8f_t acc = accs[mh][nh];
#pragma unroll
            for (int r = 0; r < 8; ++r) {
                int m = mt * 32 + mh * 16 + r + hi * 8;
                float v = acc[r] + bv;
                if (GELU) v = 0.5f * v * (1.0f + erff(v * 0.70710678118654752f));
                size_t off = (size_t)m * N + n;
                if (OUT_BF16) ((bf16_t*)out)[off] = (bf16_t)v;
                else {
                    float* o = (float*)out;
                    if (ACCUM) o[off] += v; else o[off] = v;
                }
            }
        }
    }
}

// ---------------------------------------------------------------------------
// Window gather: x_ln (B,H,W,C) bf16 -> xw (nWin*49, C) bf16 with cyclic -shift
// ---------------------------------------------------------------------------
__global__ void k_window_gather(const bf16_t* __restrict__ xln, bf16_t* __restrict__ xw,
                                int Hp, int C, int shift, long total)
{
    long t = (long)blockIdx.x * blockDim.x + threadIdx.x;
    if (t >= total) return;
    int c = (int)(t % C);
    long row = t / C;
    int i = (int)(row % 49);
    long wdx = row / 49;
    int nw_ = Hp / 7;
    int nwin = nw_ * nw_;
    int wi = (int)(wdx % nwin);
    int b  = (int)(wdx / nwin);
    int wh = wi / nw_, ww = wi % nw_;
    int r = i / 7, s = i % 7;
    int gh = wh * 7 + r + shift; if (gh >= Hp) gh -= Hp;
    int gw = ww * 7 + s + shift; if (gw >= Hp) gw -= Hp;
    xw[t] = xln[(((size_t)b * Hp + gh) * Hp + gw) * C + c];
}

// Window scatter-add (inverse mapping): h(B,H,W,C) += proj(nWin*49, C)
__global__ void k_window_scatter_add(const float* __restrict__ po, float* __restrict__ h,
                                     int Hp, int C, int shift, long total)
{
    long t = (long)blockIdx.x * blockDim.x + threadIdx.x;
    if (t >= total) return;
    int c = (int)(t % C);
    long row = t / C;
    int i = (int)(row % 49);
    long wdx = row / 49;
    int nw_ = Hp / 7;
    int nwin = nw_ * nw_;
    int wi = (int)(wdx % nwin);
    int b  = (int)(wdx / nwin);
    int wh = wi / nw_, ww = wi % nw_;
    int r = i / 7, s = i % 7;
    int gh = wh * 7 + r + shift; if (gh >= Hp) gh -= Hp;
    int gw = ww * 7 + s + shift; if (gw >= Hp) gw -= Hp;
    h[(((size_t)b * Hp + gh) * Hp + gw) * C + c] += po[t];
}

// ---------------------------------------------------------------------------
// Windowed attention: one block (256 thr = 8 waves) per (window, head).
// qkv: (nWin*49, 3C) bf16; out: (nWin*49, C) bf16; head dim = 32.
// Bias+mask precomputed into Ss and fed to WMMA as the C operand.
// ---------------------------------------------------------------------------
__device__ inline int swin_region(int g, int Hp)
{
    return (g < Hp - 7) ? 0 : ((g < Hp - 3) ? 1 : 2);
}

__global__ __launch_bounds__(256)
void k_attn(const bf16_t* __restrict__ qkv, bf16_t* __restrict__ out,
            const float* __restrict__ rpb, int nH, int C,
            int nw_, int Hp, int shift)
{
    __shared__ __align__(16) bf16_t qs[64 * 32];
    __shared__ __align__(16) bf16_t ks[64 * 32];
    __shared__ __align__(16) bf16_t vts[32 * 64];   // V transposed: vts[col*64 + row]
    __shared__ __align__(16) float  Ss[64 * 64];
    __shared__ __align__(16) bf16_t Ps[64 * 64];
    __shared__ float biasb[169];                    // rpb[:, head]
    __shared__ int   rid[49];                       // shifted-window region id

    int w = blockIdx.x;     // global window index
    int hh = blockIdx.y;    // head
    int tid = threadIdx.x;
    int C3 = 3 * C;
    size_t base = (size_t)w * 49 * C3 + (size_t)hh * 32;
    const float scale = 0.17677669529663687f;  // 1/sqrt(32)

    int nwin = nw_ * nw_;
    int wi = w % nwin;
    int wh = wi / nw_, ww = wi % nw_;

    // stage rpb column for this head + region ids
    for (int i = tid; i < 169; i += 256) biasb[i] = rpb[(size_t)i * nH + hh];
    if (tid < 49)
        rid[tid] = (shift > 0)
                 ? swin_region(wh * 7 + tid / 7, Hp) * 3 + swin_region(ww * 7 + tid % 7, Hp)
                 : 0;

    // stage q (scaled), k, v^T into LDS; 256 threads x 8 elems = 64x32
    {
        int r = tid >> 2, cb = (tid & 3) * 8;
        if (r < 49) {
            const bf16_t* p = qkv + base + (size_t)r * C3 + cb;
            v8bf_t q8 = *(const v8bf_t*)(p);
            v8bf_t k8 = *(const v8bf_t*)(p + C);
            v8bf_t v8 = *(const v8bf_t*)(p + 2 * C);
            v8bf_t qsc;
#pragma unroll
            for (int j = 0; j < 8; ++j) qsc[j] = (bf16_t)((float)q8[j] * scale);
            *(v8bf_t*)&qs[r * 32 + cb] = qsc;
            *(v8bf_t*)&ks[r * 32 + cb] = k8;
#pragma unroll
            for (int j = 0; j < 8; ++j) vts[(cb + j) * 64 + r] = v8[j];
        } else {
            v8bf_t z = {};
            *(v8bf_t*)&qs[r * 32 + cb] = z;
            *(v8bf_t*)&ks[r * 32 + cb] = z;
#pragma unroll
            for (int j = 0; j < 8; ++j) vts[(cb + j) * 64 + r] = (bf16_t)0.0f;
        }
    }
    __syncthreads();

    // precompute S init = rel-pos bias + shift mask (padded to 64x64)
    for (int idx = tid; idx < 64 * 64; idx += 256) {
        int m = idx >> 6, n = idx & 63;
        float s = 0.f;
        if (m < 49 && n < 49) {
            int rel = (m / 7 - n / 7 + 6) * 13 + (m % 7 - n % 7 + 6);
            s = biasb[rel];
            if (shift > 0 && rid[m] != rid[n]) s -= 100.0f;
        }
        Ss[idx] = s;
    }
    __syncthreads();

    int wave = tid >> 5, lane = tid & 31;
    int l15 = lane & 15, hi = lane >> 4;

    // S = q @ k^T + Sinit  (16 tiles, 2 per wave; C operand = bias+mask)
    for (int t = wave; t < 16; t += 8) {
        int mt = t >> 2, nt = t & 3;
        int m0 = mt * 16 + l15;
        int n0 = nt * 16 + l15;
        AFrag a, b;
        a.h[0] = *(const v8bf_t*)&qs[m0 * 32 + hi * 8];
        a.h[1] = *(const v8bf_t*)&qs[m0 * 32 + hi * 8 + 16];
        b.h[0] = *(const v8bf_t*)&ks[n0 * 32 + hi * 8];
        b.h[1] = *(const v8bf_t*)&ks[n0 * 32 + hi * 8 + 16];
        v8f_t cfr;
#pragma unroll
        for (int r = 0; r < 8; ++r)
            cfr[r] = Ss[(mt * 16 + r + hi * 8) * 64 + n0];
        v8f_t acc = __builtin_amdgcn_wmma_f32_16x16x32_bf16(false, a.v, false, b.v,
                                                            (short)0, cfr, false, false);
#pragma unroll
        for (int r = 0; r < 8; ++r)
            Ss[(mt * 16 + r + hi * 8) * 64 + n0] = acc[r];
    }
    __syncthreads();

    // Row softmax (rows 0..48), zero-pad rows 49..63
    if (tid < 64) {
        if (tid < 49) {
            float mx = -1e30f;
            for (int j = 0; j < 49; ++j) mx = fmaxf(mx, Ss[tid * 64 + j]);
            float sum = 0.f;
            for (int j = 0; j < 49; ++j) sum += __expf(Ss[tid * 64 + j] - mx);
            float inv = 1.0f / sum;
            for (int j = 0; j < 64; ++j) {
                float p = (j < 49) ? __expf(Ss[tid * 64 + j] - mx) * inv : 0.0f;
                Ps[tid * 64 + j] = (bf16_t)p;
            }
        } else {
            for (int j = 0; j < 64; ++j) Ps[tid * 64 + j] = (bf16_t)0.0f;
        }
    }
    __syncthreads();

    // O = P(64x64) @ V(64x32): 8 tiles, 1 per wave, K-loop of 2
    {
        int mt = wave >> 1, nt = wave & 1;
        v8f_t acc = {};
        for (int k0 = 0; k0 < 64; k0 += 32) {
            int m0 = mt * 16 + l15;
            int n0 = nt * 16 + l15;
            AFrag a, b;
            a.h[0] = *(const v8bf_t*)&Ps[m0 * 64 + k0 + hi * 8];
            a.h[1] = *(const v8bf_t*)&Ps[m0 * 64 + k0 + hi * 8 + 16];
            b.h[0] = *(const v8bf_t*)&vts[n0 * 64 + k0 + hi * 8];
            b.h[1] = *(const v8bf_t*)&vts[n0 * 64 + k0 + hi * 8 + 16];
            acc = __builtin_amdgcn_wmma_f32_16x16x32_bf16(false, a.v, false, b.v,
                                                          (short)0, acc, false, false);
        }
#pragma unroll
        for (int r = 0; r < 8; ++r) {
            int m = mt * 16 + r + hi * 8;
            int n = nt * 16 + l15;
            if (m < 49)
                out[(size_t)(w * 49 + m) * C + hh * 32 + n] = (bf16_t)acc[r];
        }
    }
}

// ---------------------------------------------------------------------------
// Patch merge gather: h(B,H,W,C) -> tmp(B*(H/2)*(W/2), 4C) f32 (x0,x1,x2,x3)
// ---------------------------------------------------------------------------
__global__ void k_merge_gather(const float* __restrict__ h, float* __restrict__ tmp,
                               int Hp, int C, long total)
{
    long t = (long)blockIdx.x * blockDim.x + threadIdx.x;
    if (t >= total) return;
    int C4 = 4 * C;
    int c4 = (int)(t % C4);
    long row = t / C4;
    int H2 = Hp / 2;
    int j = (int)(row % H2); long r2 = row / H2;
    int i = (int)(r2 % H2); int b = (int)(r2 / H2);
    int which = c4 / C, c = c4 % C;
    int di = (which == 1 || which == 3) ? 1 : 0;  // x0:(0,0) x1:(1,0) x2:(0,1) x3:(1,1)
    int dj = (which >= 2) ? 1 : 0;
    tmp[t] = h[(((size_t)b * Hp + (2 * i + di)) * Hp + (2 * j + dj)) * C + c];
}

// Mean pool over HW: x(B,HW,C) f32 -> out(B,C) bf16
__global__ void k_mean_pool(const float* __restrict__ x, bf16_t* __restrict__ out,
                            int HW, int C, int total)
{
    int t = blockIdx.x * blockDim.x + threadIdx.x;
    if (t >= total) return;
    int c = t % C, b = t / C;
    const float* p = x + (size_t)b * HW * C + c;
    float s = 0.f;
    for (int i = 0; i < HW; ++i) s += p[(size_t)i * C];
    out[t] = (bf16_t)(s / HW);
}

// Final 128->1 head: out[b] = h1[b,:] . w2 + b2
__global__ void k_final_head(const float* __restrict__ h1, const float* __restrict__ w2,
                             const float* __restrict__ b2, float* __restrict__ out, int B)
{
    int b = blockIdx.x * blockDim.x + threadIdx.x;
    if (b >= B) return;
    float s = b2[0];
    for (int j = 0; j < 128; ++j) s += h1[b * 128 + j] * w2[j];
    out[b] = s;
}

// ---------------------------------------------------------------------------
// Host orchestration
// ---------------------------------------------------------------------------
struct BlockP {
    const float *fc1b, *fc1w, *fc2b, *fc2w, *n1b, *n1w, *n2b, *n2w,
                *projb, *projw, *qkvb, *qkvw, *rpb;
};
struct MergeP { const float *nb, *nw, *rw; };

extern "C" void kernel_launch(void* const* d_in, const int* in_sizes, int n_in,
                              void* d_out, int out_size, void* d_ws, size_t ws_size,
                              hipStream_t stream)
{
    (void)out_size;
    static const int DEPTHS[4] = {2, 2, 6, 2};
    static const int DIMS[4]   = {96, 192, 384, 768};
    static const int NHEADS[4] = {3, 6, 12, 24};
    const int B = 32;
    const size_t MB = 1024ull * 1024ull;
    if (ws_size < 280 * MB) return;  // workspace budget (see layout below)

    // ---- locate inputs: x identified by its unique size; params flattened in
    // ---- jax pytree order (dict keys sorted alphabetically, lists in order).
    const long XSZ = (long)B * 3 * 224 * 224;
    int xidx = (in_sizes[0] == XSZ) ? 0 : (n_in - 1);
    const float* x = (const float*)d_in[xidx];
    int pi = (xidx == 0) ? 1 : 0;
    auto F = [&]() { return (const float*)d_in[pi++]; };

    const float* b1    = F();
    const float* b2    = F();
    const float* normb = F();
    const float* normw = F();
    const float* patch_b  = F();
    const float* patch_nb = F();
    const float* patch_nw = F();
    const float* patch_w  = F();
    BlockP blocks[4][6];
    MergeP merges[3];
    for (int s = 0; s < 4; ++s) {
        for (int d = 0; d < DEPTHS[s]; ++d) {
            BlockP& bp = blocks[s][d];
            bp.fc1b = F(); bp.fc1w = F(); bp.fc2b = F(); bp.fc2w = F();
            bp.n1b = F();  bp.n1w = F();  bp.n2b = F();  bp.n2w = F();
            bp.projb = F(); bp.projw = F(); bp.qkvb = F(); bp.qkvw = F();
            bp.rpb = F();
        }
        if (s < 3) { merges[s].nb = F(); merges[s].nw = F(); merges[s].rw = F(); }
    }
    const float* w1 = F();
    const float* w2 = F();

    // ---- workspace layout (bytes) ----
    char* ws = (char*)d_ws;
    float*  h_buf   = (float*) (ws + 0);        // residual stream (B,H,W,C) f32  <=38.6MB
    bf16_t* xln_buf = (bf16_t*)(ws + 40 * MB);  // LN output / pooled, bf16       <=19.3MB
    bf16_t* xw_buf  = (bf16_t*)(ws + 60 * MB);  // windowed LN, bf16              <=19.3MB
    bf16_t* att_buf = (bf16_t*)(ws + 80 * MB);  // attention output, bf16         <=19.3MB
    bf16_t* big_buf = (bf16_t*)(ws + 100 * MB); // qkv / fc1 output, bf16         <=77.1MB
    float*  tmp_buf = (float*) (ws + 180 * MB); // proj out / merge / final, f32  <=38.6MB
    bf16_t* wb      = (bf16_t*)(ws + 220 * MB); // bf16 weight arena              <=55MB
    size_t wboff = 0;
    auto CONV = [&](const float* src, long n) -> bf16_t* {
        wboff = (wboff + 7) & ~(size_t)7;       // keep 16B alignment for b128 loads
        bf16_t* dst = wb + wboff;
        wboff += (size_t)n;
        k_f32_to_bf16<<<dim3((unsigned)((n + 255) / 256)), dim3(256), 0, stream>>>(src, dst, n);
        return dst;
    };

    bf16_t *qkvw_bf[4][6], *projw_bf[4][6], *fc1w_bf[4][6], *fc2w_bf[4][6], *rw_bf[3];
    for (int s = 0; s < 4; ++s) {
        long C = DIMS[s];
        for (int d = 0; d < DEPTHS[s]; ++d) {
            qkvw_bf[s][d] = CONV(blocks[s][d].qkvw, 3 * C * C);
            projw_bf[s][d] = CONV(blocks[s][d].projw, C * C);
            fc1w_bf[s][d] = CONV(blocks[s][d].fc1w, 4 * C * C);
            fc2w_bf[s][d] = CONV(blocks[s][d].fc2w, 4 * C * C);
        }
        if (s < 3) rw_bf[s] = CONV(merges[s].rw, 8 * C * C);
    }
    bf16_t* w1_bf = CONV(w1, 128 * 768);

    // ---- patch embed + norm ----
    {
        long total = (long)B * 56 * 56 * 96;
        k_patch_embed<<<dim3((unsigned)((total + 255) / 256)), dim3(256), 0, stream>>>(
            x, patch_w, patch_b, h_buf, total);
        k_layernorm<false><<<dim3((unsigned)(B * 56 * 56)), dim3(128), 0, stream>>>(
            h_buf, patch_nw, patch_nb, h_buf, 96);
    }

    // ---- stages ----
    int Hp = 56;
    for (int s = 0; s < 4; ++s) {
        int C = DIMS[s], nH = NHEADS[s];
        int rows = B * Hp * Hp;       // also M (= nWin*49)
        int nw_ = Hp / 7;
        int nWin = B * nw_ * nw_;
        int M = rows;
        for (int d = 0; d < DEPTHS[s]; ++d) {
            BlockP& bp = blocks[s][d];
            int shift = ((d & 1) && Hp > 7) ? 3 : 0;
            long tot = (long)M * C;
            unsigned eg = (unsigned)((tot + 255) / 256);
            unsigned mg = (unsigned)((M + 127) / 128);

            // attention half
            k_layernorm<true><<<dim3((unsigned)rows), dim3(128), 0, stream>>>(
                h_buf, bp.n1w, bp.n1b, xln_buf, C);
            k_window_gather<<<dim3(eg), dim3(256), 0, stream>>>(
                xln_buf, xw_buf, Hp, C, shift, tot);
            k_gemm<true, false, false><<<dim3(3 * C / 32, mg), dim3(128), 0, stream>>>(
                xw_buf, qkvw_bf[s][d], bp.qkvb, big_buf, M, 3 * C, C);
            k_attn<<<dim3((unsigned)nWin, (unsigned)nH), dim3(256), 0, stream>>>(
                big_buf, att_buf, bp.rpb, nH, C, nw_, Hp, shift);
            k_gemm<false, false, false><<<dim3(C / 32, mg), dim3(128), 0, stream>>>(
                att_buf, projw_bf[s][d], bp.projb, tmp_buf, M, C, C);
            k_window_scatter_add<<<dim3(eg), dim3(256), 0, stream>>>(
                tmp_buf, h_buf, Hp, C, shift, tot);

            // MLP half
            k_layernorm<true><<<dim3((unsigned)rows), dim3(128), 0, stream>>>(
                h_buf, bp.n2w, bp.n2b, xln_buf, C);
            k_gemm<true, true, false><<<dim3(4 * C / 32, mg), dim3(128), 0, stream>>>(
                xln_buf, fc1w_bf[s][d], bp.fc1b, big_buf, M, 4 * C, C);
            k_gemm<false, false, true><<<dim3(C / 32, mg), dim3(128), 0, stream>>>(
                big_buf, fc2w_bf[s][d], bp.fc2b, h_buf, M, C, 4 * C);
        }
        if (s < 3) {
            int H2 = Hp / 2;
            int M3 = B * H2 * H2;
            long tot = (long)M3 * 4 * C;
            k_merge_gather<<<dim3((unsigned)((tot + 255) / 256)), dim3(256), 0, stream>>>(
                h_buf, tmp_buf, Hp, C, tot);
            k_layernorm<true><<<dim3((unsigned)M3), dim3(128), 0, stream>>>(
                tmp_buf, merges[s].nw, merges[s].nb, xln_buf, 4 * C);
            k_gemm<false, false, false><<<dim3(2 * C / 32, (unsigned)((M3 + 127) / 128)),
                                          dim3(128), 0, stream>>>(
                xln_buf, rw_bf[s], nullptr, h_buf, M3, 2 * C, 4 * C);
            Hp = H2;
        }
    }

    // ---- final norm, pool, heads ----
    {
        int rows = B * 7 * 7;
        k_layernorm<false><<<dim3((unsigned)rows), dim3(128), 0, stream>>>(
            h_buf, normw, normb, tmp_buf, 768);
        k_mean_pool<<<dim3((unsigned)((B * 768 + 255) / 256)), dim3(256), 0, stream>>>(
            tmp_buf, xln_buf, 49, 768, B * 768);
        k_gemm<false, false, false><<<dim3(128 / 32, 1), dim3(128), 0, stream>>>(
            xln_buf, w1_bf, b1, tmp_buf, 32, 128, 768);
        k_final_head<<<dim3(1), dim3(32), 0, stream>>>(
            tmp_buf, w2, b2, (float*)d_out, B);
    }
}
